// VectorQuantizerEMA_49082886259004
// MI455X (gfx1250) — matrix operand order repaired and checked by reading, not compiled, not run on hardware
//
#include <hip/hip_runtime.h>
#include <hip/hip_bf16.h>

typedef __attribute__((ext_vector_type(16))) __bf16 v16bf;
typedef __attribute__((ext_vector_type(8)))  float  v8f;

#define NTOK 32768
#define KEMB 8192
#define DDIM 512

#define BM 128
#define BN 128
#define BKE 64      // bf16 K elements staged per chunk (two wmma k-steps)
#define LDSTR 72    // padded LDS row stride in bf16 (36 dwords: conflict-free, 16B aligned)
#define PLANE (BM * LDSTR)            // bf16 elems per LDS plane
#define NCH ((KEMB / BN) * (DDIM / BKE))   // 64 * 8 = 512 pipeline chunks
#define SMEM_BYTES (8 * PLANE * 2 + BM * 8)  // 2 buffers x 4 planes + rowkey

#define DECAY 0.99f
#define EPSV 1e-05f
#define CCOST 0.25f

__device__ __forceinline__ unsigned long long shflxor_u64(unsigned long long v, int m) {
    unsigned int lo = (unsigned int)(v & 0xffffffffull);
    unsigned int hi = (unsigned int)(v >> 32);
    lo = __shfl_xor(lo, m, 32);
    hi = __shfl_xor(hi, m, 32);
    return (((unsigned long long)hi) << 32) | (unsigned long long)lo;
}

// Monotonic float -> uint mapping (total order preserved in packed key)
__device__ __forceinline__ unsigned int order_f32(float f) {
    unsigned int u = __float_as_uint(f);
    return (u & 0x80000000u) ? ~u : (u | 0x80000000u);
}

// CDNA5 async global->LDS copy (16B per lane), tracked on ASYNCcnt.
__device__ __forceinline__ void async_cp16(void* lds, const void* gsrc) {
    unsigned dst = (unsigned)(unsigned long long)lds;
    unsigned long long src = (unsigned long long)gsrc;
    asm volatile("global_load_async_to_lds_b128 %0, %1, off"
                 :: "v"(dst), "v"(src) : "memory");
}
// Async loads complete in issue order: waiting <=16 means the previous
// stage's 16 copies (per wave) have landed.
__device__ __forceinline__ void wait_async_le16() {
    asm volatile("s_wait_asynccnt 0x10" ::: "memory");
}
__device__ __forceinline__ void wait_async_le0() {
    asm volatile("s_wait_asynccnt 0x0" ::: "memory");
}

// ---------------------------------------------------------------------------
// 0) split fp32 -> bf16 hi/lo planes (bf16x3 GEMM emulation prepass)
// ---------------------------------------------------------------------------
__global__ __launch_bounds__(256) void vq_split(const float* __restrict__ src,
                                                __bf16* __restrict__ hi,
                                                __bf16* __restrict__ lo, int n) {
    int t = blockIdx.x * blockDim.x + threadIdx.x;
    if (t >= n) return;
    float x = src[t];
    __bf16 h = (__bf16)x;
    hi[t] = h;
    lo[t] = (__bf16)(x - (float)h);
}

// ---------------------------------------------------------------------------
// 1) init: new_ema_w_out = DECAY * ema_w, zero counts + scalar accumulators
// ---------------------------------------------------------------------------
__global__ __launch_bounds__(256) void vq_init(const float* __restrict__ ema_w,
                                               float* __restrict__ emaw_out,
                                               float* __restrict__ counts,
                                               float* __restrict__ scalars) {
    int t = blockIdx.x * blockDim.x + threadIdx.x;
    if (t < KEMB * DDIM) emaw_out[t] = DECAY * ema_w[t];
    if (t < KEMB) counts[t] = 0.0f;
    if (t < 4) scalars[t] = 0.0f;
}

// ---------------------------------------------------------------------------
// 2) |e_k|^2 row norms (fp32)
// ---------------------------------------------------------------------------
__global__ __launch_bounds__(128) void vq_row_norms(const float* __restrict__ E,
                                                    float* __restrict__ enorm) {
    const int k = blockIdx.x;
    const float4 v = ((const float4*)(E + (size_t)k * DDIM))[threadIdx.x];
    float s = v.x * v.x + v.y * v.y + v.z * v.z + v.w * v.w;
    #pragma unroll
    for (int m = 16; m > 0; m >>= 1) s += __shfl_xor(s, m, 32);
    __shared__ float red[4];
    if ((threadIdx.x & 31) == 0) red[threadIdx.x >> 5] = s;
    __syncthreads();
    if (threadIdx.x == 0) enorm[k] = red[0] + red[1] + red[2] + red[3];
}

// ---------------------------------------------------------------------------
// 3) Fused bf16x3 GEMM (v_wmma_f32_16x16x32_bf16) + per-row argmin over K.
//    256 threads (8 waves), 128x128 tile, D chunked by 64 (2 k-steps).
//    Double-buffered async global->LDS staging pipelined against WMMA.
// ---------------------------------------------------------------------------
__global__ __launch_bounds__(256) void vq_dist_argmin(const __bf16* __restrict__ Xh,
                                                      const __bf16* __restrict__ Xl,
                                                      const __bf16* __restrict__ Eh,
                                                      const __bf16* __restrict__ El,
                                                      const float* __restrict__ enorm,
                                                      int* __restrict__ idx_out) {
    extern __shared__ __align__(16) __bf16 smem[];
    unsigned long long* rowkey = (unsigned long long*)(smem + 8 * PLANE);

    const int tid  = threadIdx.x;
    const int wave = tid >> 5;
    const int lane = tid & 31;
    const int half = lane >> 4;        // 0: lanes 0-15, 1: lanes 16-31
    const int lm   = lane & 15;

    const int rowBase = blockIdx.x * BM;
    const int wr = (wave >> 1) * 32;   // wave row offset (0/32/64/96)
    const int wc = (wave & 1) * 64;    // wave col offset (0/64)

    // staging map: 8 threads per row (8 bf16 = 16B each), 32 rows/pass, 4 passes
    const int sr = tid >> 3;           // 0..31
    const int sc = (tid & 7) * 8;      // 0..56

    if (tid < BM) rowkey[tid] = 0xFFFFFFFFFFFFFFFFull;

    // issue 16 async copies for pipeline chunk cc into buffer buf
    auto issue_stage = [&](int cc, int buf) {
        const int ctb = (cc >> 3) * BN;
        const int dk  = (cc & 7) * BKE;
        __bf16* dAh = smem + buf * 4 * PLANE;
        __bf16* dAl = dAh + PLANE;
        __bf16* dBh = dAl + PLANE;
        __bf16* dBl = dBh + PLANE;
        #pragma unroll
        for (int p = 0; p < 4; ++p) {
            const int r = sr + p * 32;
            const size_t ga = (size_t)(rowBase + r) * DDIM + dk + sc;
            const size_t gb = (size_t)(ctb + r) * DDIM + dk + sc;
            const int lo = r * LDSTR + sc;
            async_cp16(dAh + lo, Xh + ga);
            async_cp16(dAl + lo, Xl + ga);
            async_cp16(dBh + lo, Eh + gb);
            async_cp16(dBl + lo, El + gb);
        }
    };

    v8f acc[2][4] = {};
    issue_stage(0, 0);

    for (int c = 0; c < NCH; ++c) {
        const int buf = c & 1;
        if ((c & 7) == 0) {
            #pragma unroll
            for (int i = 0; i < 2; ++i)
                #pragma unroll
                for (int j = 0; j < 4; ++j) acc[i][j] = (v8f){};
        }
        if (c + 1 < NCH) { issue_stage(c + 1, buf ^ 1); wait_async_le16(); }
        else             { wait_async_le0(); }
        __syncthreads();   // current buffer visible to all waves

        const __bf16* Ahp = smem + buf * 4 * PLANE;
        const __bf16* Alp = Ahp + PLANE;
        const __bf16* Bhp = Alp + PLANE;
        const __bf16* Blp = Bhp + PLANE;

        #pragma unroll
        for (int ks = 0; ks < 2; ++ks) {
            const int k0 = ks * 32;
            v16bf ah[2], alo[2], bh[4], blo[4];
            #pragma unroll
            for (int i = 0; i < 2; ++i) {
                const int row = (wr + i * 16 + lm) * LDSTR + k0;
                ((float4*)&ah[i])[0]  = *(const float4*)(Ahp + row + half * 8);
                ((float4*)&ah[i])[1]  = *(const float4*)(Ahp + row + 16 + half * 8);
                ((float4*)&alo[i])[0] = *(const float4*)(Alp + row + half * 8);
                ((float4*)&alo[i])[1] = *(const float4*)(Alp + row + 16 + half * 8);
            }
            #pragma unroll
            for (int j = 0; j < 4; ++j) {
                const int col = (wc + j * 16 + lm) * LDSTR + k0;
                ((float4*)&bh[j])[0]  = *(const float4*)(Bhp + col + half * 8);
                ((float4*)&bh[j])[1]  = *(const float4*)(Bhp + col + 16 + half * 8);
                ((float4*)&blo[j])[0] = *(const float4*)(Blp + col + half * 8);
                ((float4*)&blo[j])[1] = *(const float4*)(Blp + col + 16 + half * 8);
            }
            #pragma unroll
            for (int i = 0; i < 2; ++i) {
                #pragma unroll
                for (int j = 0; j < 4; ++j) {
                    acc[i][j] = __builtin_amdgcn_wmma_f32_16x16x32_bf16(
                        false, ah[i], false, bh[j], (short)0, acc[i][j], false, false);
                    acc[i][j] = __builtin_amdgcn_wmma_f32_16x16x32_bf16(
                        false, ah[i], false, blo[j], (short)0, acc[i][j], false, false);
                    acc[i][j] = __builtin_amdgcn_wmma_f32_16x16x32_bf16(
                        false, alo[i], false, bh[j], (short)0, acc[i][j], false, false);
                }
            }
        }

        if ((c & 7) == 7) {
            // epilogue: dist = |e|^2 - 2*dot; per-row argmin (|x|^2 irrelevant)
            const int ctBase = (c >> 3) * BN;
            float en[4];
            #pragma unroll
            for (int j = 0; j < 4; ++j) en[j] = enorm[ctBase + wc + j * 16 + lm];

            #pragma unroll
            for (int i = 0; i < 2; ++i) {
                #pragma unroll
                for (int r = 0; r < 8; ++r) {
                    unsigned long long key = 0xFFFFFFFFFFFFFFFFull;
                    #pragma unroll
                    for (int j = 0; j < 4; ++j) {
                        const int gcol = ctBase + wc + j * 16 + lm;
                        const float dist = en[j] - 2.0f * acc[i][j][r];
                        unsigned long long k2 =
                            (((unsigned long long)order_f32(dist)) << 32) |
                            (unsigned long long)(unsigned int)gcol;
                        key = (k2 < key) ? k2 : key;
                    }
                    #pragma unroll
                    for (int m = 1; m < 16; m <<= 1) {
                        unsigned long long o = shflxor_u64(key, m);
                        key = (o < key) ? o : key;
                    }
                    if (lm == 0) {
                        atomicMin(&rowkey[wr + i * 16 + half * 8 + r], key);
                    }
                }
            }
        }
        __syncthreads();   // all waves done reading buf before it is re-filled
    }

    if (tid < BM) idx_out[rowBase + tid] = (int)(rowkey[tid] & 0xFFFFFFFFull);
}

// ---------------------------------------------------------------------------
// 4) gather quantized rows (quantized_st == quantized numerically), loss,
//    histogram, dw scatter into pre-scaled new_ema_w
// ---------------------------------------------------------------------------
__global__ __launch_bounds__(128) void vq_scatter(const float* __restrict__ X,
                                                  const float* __restrict__ E,
                                                  const int* __restrict__ idx,
                                                  float* __restrict__ q_out,
                                                  float* __restrict__ emaw_out,
                                                  float* __restrict__ counts,
                                                  float* __restrict__ scalars) {
    const int i = blockIdx.x;
    const int t = threadIdx.x;           // 128 threads x float4 = 512 elems
    const int k = idx[i];

    const float4 x = ((const float4*)(X + (size_t)i * DDIM))[t];
    const float4 e = ((const float4*)(E + (size_t)k * DDIM))[t];
    ((float4*)(q_out + (size_t)i * DDIM))[t] = e;

    const float dx = e.x - x.x, dy = e.y - x.y, dz = e.z - x.z, dw = e.w - x.w;
    float s = dx * dx + dy * dy + dz * dz + dw * dw;

    float* wrow = emaw_out + (size_t)k * DDIM + t * 4;
    atomicAdd(wrow + 0, (1.0f - DECAY) * x.x);
    atomicAdd(wrow + 1, (1.0f - DECAY) * x.y);
    atomicAdd(wrow + 2, (1.0f - DECAY) * x.z);
    atomicAdd(wrow + 3, (1.0f - DECAY) * x.w);

    #pragma unroll
    for (int m = 16; m > 0; m >>= 1) s += __shfl_xor(s, m, 32);
    __shared__ float red[4];
    if ((t & 31) == 0) red[t >> 5] = s;
    __syncthreads();
    if (t == 0) {
        atomicAdd(&scalars[0], red[0] + red[1] + red[2] + red[3]);  // loss sum
        atomicAdd(&counts[k], 1.0f);
    }
}

// ---------------------------------------------------------------------------
// 5) EMA cluster size pre-normalization + n_total + entropy reductions
// ---------------------------------------------------------------------------
__global__ __launch_bounds__(256) void vq_finalize_a(const float* __restrict__ ema_cs,
                                                     const float* __restrict__ counts,
                                                     float* __restrict__ pre,
                                                     float* __restrict__ scalars) {
    int k = blockIdx.x * blockDim.x + threadIdx.x;
    if (k >= KEMB) return;
    float p = ema_cs[k] * DECAY + (1.0f - DECAY) * counts[k];
    pre[k] = p;
    atomicAdd(&scalars[1], p);                         // n_total
    float ap = counts[k] / (float)NTOK;
    atomicAdd(&scalars[2], ap * logf(ap + 1e-10f));    // entropy sum
}

// ---------------------------------------------------------------------------
// 6) new_cs + scalar outputs (loss, perplexity)
// ---------------------------------------------------------------------------
__global__ __launch_bounds__(256) void vq_finalize_b(const float* __restrict__ pre,
                                                     float* __restrict__ ncs_ws,
                                                     float* __restrict__ ncs_out,
                                                     float* __restrict__ loss_out,
                                                     float* __restrict__ perp_out,
                                                     const float* __restrict__ scalars) {
    int k = blockIdx.x * blockDim.x + threadIdx.x;
    if (k >= KEMB) return;
    const float nt = scalars[1];
    const float v = (pre[k] + EPSV) / (nt + (float)KEMB * EPSV) * nt;
    ncs_ws[k] = v;
    ncs_out[k] = v;
    if (k == 0) {
        *loss_out = CCOST * scalars[0] / (float)((size_t)NTOK * DDIM);
        *perp_out = expf(-scalars[2]);
    }
}

// ---------------------------------------------------------------------------
// 7) new_embedding = new_ema_w / new_cs[:, None]
// ---------------------------------------------------------------------------
__global__ __launch_bounds__(256) void vq_finalize_c(const float* __restrict__ emaw_out,
                                                     const float* __restrict__ ncs_ws,
                                                     float* __restrict__ emb_out) {
    int t = blockIdx.x * blockDim.x + threadIdx.x;
    if (t >= KEMB * DDIM) return;
    emb_out[t] = emaw_out[t] / ncs_ws[t / DDIM];
}

// ---------------------------------------------------------------------------
extern "C" void kernel_launch(void* const* d_in, const int* in_sizes, int n_in,
                              void* d_out, int out_size, void* d_ws, size_t ws_size,
                              hipStream_t stream) {
    const float* X      = (const float*)d_in[0];  // [N, D]
    const float* E      = (const float*)d_in[1];  // [K, D]
    const float* ema_cs = (const float*)d_in[2];  // [K]
    const float* ema_w  = (const float*)d_in[3];  // [K, D]

    float* out      = (float*)d_out;
    float* q_out    = out;                                   // N*D
    float* loss_out = out + (size_t)NTOK * DDIM;             // 1
    float* perp_out = loss_out + 1;                          // 1
    float* emb_out  = perp_out + 1;                          // K*D
    float* ncs_out  = emb_out + (size_t)KEMB * DDIM;         // K
    float* emaw_out = ncs_out + KEMB;                        // K*D

    const size_t ND = (size_t)NTOK * DDIM;
    const size_t KD = (size_t)KEMB * DDIM;

    __bf16* Xh = (__bf16*)d_ws;           // bf16 hi/lo planes (prepass output)
    __bf16* Xl = Xh + ND;
    __bf16* Eh = Xl + ND;
    __bf16* El = Eh + KD;
    int*   idx     = (int*)(El + KD);                        // N ints
    float* enorm   = (float*)(idx + NTOK);                   // K
    float* counts  = enorm + KEMB;                           // K
    float* pre     = counts + KEMB;                          // K
    float* ncs     = pre + KEMB;                             // K
    float* scalars = ncs + KEMB;                             // 4

    vq_split<<<(int)((ND + 255) / 256), 256, 0, stream>>>(X, Xh, Xl, (int)ND);
    vq_split<<<(int)((KD + 255) / 256), 256, 0, stream>>>(E, Eh, El, (int)KD);
    vq_init<<<(int)((KD + 255) / 256), 256, 0, stream>>>(ema_w, emaw_out, counts, scalars);
    vq_row_norms<<<KEMB, 128, 0, stream>>>(E, enorm);
    vq_dist_argmin<<<NTOK / BM, 256, SMEM_BYTES, stream>>>(Xh, Xl, Eh, El, enorm, idx);
    vq_scatter<<<NTOK, 128, 0, stream>>>(X, E, idx, q_out, emaw_out, counts, scalars);
    vq_finalize_a<<<(KEMB + 255) / 256, 256, 0, stream>>>(ema_cs, counts, pre, scalars);
    vq_finalize_b<<<(KEMB + 255) / 256, 256, 0, stream>>>(pre, ncs, ncs_out, loss_out, perp_out, scalars);
    vq_finalize_c<<<(int)((KD + 255) / 256), 256, 0, stream>>>(emaw_out, ncs, emb_out);
}
